// GPT_61220463837741
// MI455X (gfx1250) — matrix operand order repaired and checked by reading, not compiled, not run on hardware
//
#include <hip/hip_runtime.h>

// ---------------------------------------------------------------------------
// GPT-2 small forward (B=4, S=1024, D=768, H=12, HS=64, L=12, V=50257)
// CDNA5 / gfx1250: wave32, WMMA f32_16x16x32_f16 for every matmul.
// Weights are transposed+converted to f16 [N,K] per layer (cheap: ~14MB/layer)
// so all A and B fragments are 16-byte vectorized loads; V is written
// pre-transposed by its projection GEMM so attention needs no gathers.
// ---------------------------------------------------------------------------

#define NV   50257
#define NVP  50432          // NV rounded up to 256 (padded f16 wte buffer)
#define ND   768
#define NS   1024
#define NH   12
#define NHS  64
#define NL   12
#define NB   4
#define NM   (NB * NS)      // 4096 rows
#define NFF  (4 * ND)       // 3072

typedef _Float16 v16h __attribute__((ext_vector_type(16)));
typedef _Float16 v8h  __attribute__((ext_vector_type(8)));
typedef float    v8f  __attribute__((ext_vector_type(8)));

enum { EPI_F16 = 0, EPI_F16T = 1, EPI_GELU_F16 = 2, EPI_RES_F32 = 3, EPI_F32 = 4 };

static __device__ __forceinline__ v8f wmma32(v16h a, v16h b, v8f c) {
    return __builtin_amdgcn_wmma_f32_16x16x32_f16(false, a, false, b,
                                                  (short)0, c, false, false);
}

static __device__ __forceinline__ v8f vzero8() {
    v8f z;
#pragma unroll
    for (int i = 0; i < 8; ++i) z[i] = 0.0f;
    return z;
}

// A fragment (16x32 f16): lane<16 rows M=lane, K = 0..15 contiguous
// (+16 shift for lanes 16..31). Two 16-byte vector loads per lane.
static __device__ __forceinline__ v16h load_a_f16(const _Float16* __restrict__ A,
                                                  int lda, int m0, int k0, int lane) {
    int row = m0 + (lane & 15);
    int kb  = (lane & 16) ? 16 : 0;
    const v8h* p = reinterpret_cast<const v8h*>(A + (size_t)row * lda + k0 + kb);
    v8h lo = p[0];
    v8h hi = p[1];              // +8 halves
    v16h f;
#pragma unroll
    for (int j = 0; j < 8; ++j) { f[j] = lo[j]; f[8 + j] = hi[j]; }
    return f;
}

// B fragment from f16 stored transposed ([N x K] row-major): lane holds
// column n0+(lane&15), K = kb..kb+15 contiguous -> two 16-byte vector loads.
static __device__ __forceinline__ v16h load_bT_f16(const _Float16* __restrict__ Bm,
                                                   int ldb, int n0, int k0, int lane) {
    int col = n0 + (lane & 15);
    int kb  = (lane & 16) ? 16 : 0;
    const v8h* p = reinterpret_cast<const v8h*>(Bm + (size_t)col * ldb + k0 + kb);
    v8h lo = p[0];
    v8h hi = p[1];
    v16h f;
#pragma unroll
    for (int j = 0; j < 8; ++j) { f[j] = lo[j]; f[8 + j] = hi[j]; }
    return f;
}

// ---------------------------------------------------------------------------
// Weight transpose + f32->f16 convert: in is [K x N] fp32 (or GPT-2 QKV
// layout [H][K][64] when QKV=1); out is [N x K] f16. 32x32 LDS tiles.
// ---------------------------------------------------------------------------
template <int QKV>
__global__ __launch_bounds__(256) void transpose_w_kernel(
    const float* __restrict__ in, _Float16* __restrict__ out, int K, int N) {
    __shared__ float tile[32][33];
    int tx = threadIdx.x & 31;
    int ty = threadIdx.x >> 5;          // 0..7
    int kt = blockIdx.x * 32;
    int nt = blockIdx.y * 32;
#pragma unroll
    for (int r = 0; r < 4; ++r) {
        int k = kt + ty + r * 8;
        int n = nt + tx;
        float v;
        if (QKV) v = in[(size_t)((n >> 6) * K + k) * NHS + (n & 63)];
        else     v = in[(size_t)k * N + n];
        tile[ty + r * 8][tx] = v;
    }
    __syncthreads();
#pragma unroll
    for (int r = 0; r < 4; ++r) {
        int n = nt + ty + r * 8;
        int k = kt + tx;
        out[(size_t)n * K + k] = (_Float16)tile[tx][ty + r * 8];
    }
}

// wte f32 -> f16 elementwise (already [V x D] = [N x K] for the tied head).
__global__ __launch_bounds__(256) void f32_to_f16_kernel(
    const float* __restrict__ in, _Float16* __restrict__ out, long n) {
    long i = (long)blockIdx.x * 1024 + threadIdx.x;
#pragma unroll
    for (int r = 0; r < 4; ++r) {
        long j = i + (long)r * 256;
        if (j < n) out[j] = (_Float16)in[j];
    }
}

// ---------------------------------------------------------------------------
// GEMM: C[M,N] = A_f16[M,K] @ Bt_f16[N,K]^T  (+bias, epilogue variants)
// Block = 256 threads = 8 waves; wave tile 32x64; block tile 64x256.
// All fragment loads are batched 16B vector loads (clause-friendly).
// ---------------------------------------------------------------------------
template <int EPI>
__global__ __launch_bounds__(256) void gemm_kernel(
    const _Float16* __restrict__ A, int lda,
    const _Float16* __restrict__ Bt,
    const float* __restrict__ bias,
    const float* __restrict__ resid,
    float* __restrict__ Of32,
    _Float16* __restrict__ Of16, int ldo,
    int M, int N, int K) {
    int lane = threadIdx.x & 31;
    int wave = threadIdx.x >> 5;
    int m0 = blockIdx.x * 64 + (wave >> 2) * 32;
    int n0 = blockIdx.y * 256 + (wave & 3) * 64;
    if (n0 >= N || m0 >= M) return;   // uniform per wave

    v8f acc[2][4];
#pragma unroll
    for (int mi = 0; mi < 2; ++mi)
#pragma unroll
        for (int ni = 0; ni < 4; ++ni) acc[mi][ni] = vzero8();

    for (int k0 = 0; k0 < K; k0 += 32) {
        v16h a0 = load_a_f16(A, lda, m0, k0, lane);
        v16h a1 = load_a_f16(A, lda, m0 + 16, k0, lane);
        v16h bf[4];
#pragma unroll
        for (int ni = 0; ni < 4; ++ni)
            bf[ni] = load_bT_f16(Bt, K, n0 + 16 * ni, k0, lane);
#pragma unroll
        for (int ni = 0; ni < 4; ++ni) {
            acc[0][ni] = wmma32(a0, bf[ni], acc[0][ni]);
            acc[1][ni] = wmma32(a1, bf[ni], acc[1][ni]);
        }
    }

    int rsel = (lane & 16) ? 8 : 0;
    int csel = lane & 15;
#pragma unroll
    for (int mi = 0; mi < 2; ++mi) {
#pragma unroll
        for (int ni = 0; ni < 4; ++ni) {
#pragma unroll
            for (int r = 0; r < 8; ++r) {
                int row = m0 + 16 * mi + rsel + r;
                int col = n0 + 16 * ni + csel;
                if (col >= N) continue;
                float v = acc[mi][ni][r];
                if (EPI != EPI_F32 && bias) v += bias[col];
                if (EPI == EPI_F16) {
                    Of16[(size_t)row * ldo + col] = (_Float16)v;
                } else if (EPI == EPI_F16T) {
                    // V projection: store transposed per (b,h): vT[b][h][e][s]
                    int bb = row >> 10, ss = row & (NS - 1);
                    size_t io = ((size_t)(bb * NH + (col >> 6)) * NHS + (col & 63)) * NS + ss;
                    Of16[io] = (_Float16)v;
                } else if (EPI == EPI_GELU_F16) {
                    float g = 0.5f * v * (1.0f + erff(v * 0.70710678118654752f));
                    Of16[(size_t)row * ldo + col] = (_Float16)g;
                } else if (EPI == EPI_RES_F32) {
                    Of32[(size_t)row * ldo + col] = resid[(size_t)row * ldo + col] + v;
                } else {  // EPI_F32 (logits)
                    Of32[(size_t)row * ldo + col] = v;
                }
            }
        }
    }
}

// ---------------------------------------------------------------------------
// Attention: causal flash-attention per (b, h). One wave owns 16 queries.
// S^T tiles (rows = key s, cols = query t) via WMMA; per-query reductions use
// one shfl_xor(16); P^T repacked into a B fragment via the same shuffles;
// O^T accumulated via WMMA with A = V^T (pre-transposed, vectorized loads).
// All loads of an s-step are issued as one batch before any compute.
// ---------------------------------------------------------------------------
__global__ __launch_bounds__(256) void attn_kernel(
    const _Float16* __restrict__ q, const _Float16* __restrict__ k,
    const _Float16* __restrict__ vt, _Float16* __restrict__ o) {
    int lane = threadIdx.x & 31;
    int wave = threadIdx.x >> 5;
    int bid  = blockIdx.x;
    int tblk = bid & 7;        // S / 128 = 8 tile-blocks
    int bh   = bid >> 3;       // 0..47
    int b    = bh / NH;
    int hh   = bh % NH;
    int hbase = hh * NHS;
    size_t rowbase = (size_t)b * NS * ND;
    int t0 = tblk * 128 + wave * 16;
    int tcol = t0 + (lane & 15);
    int rsel = (lane & 16) ? 8 : 0;

    const _Float16* qb  = q + rowbase + hbase;
    const _Float16* kb  = k + rowbase + hbase;
    const _Float16* vtb = vt + (size_t)(b * NH + hh) * NHS * NS;  // [64 e][1024 s]

    // Q^T B-fragments are loop-invariant: hoist.
    v16h qf0 = load_bT_f16(qb, ND, t0, 0, lane);
    v16h qf1 = load_bT_f16(qb, ND, t0, 32, lane);

    float mrun = -1e30f, lrun = 0.0f;
    v8f oacc[4];
#pragma unroll
    for (int et = 0; et < 4; ++et) oacc[et] = vzero8();

    for (int s0 = 0; s0 < t0 + 16; s0 += 32) {
        // ---- batch all loads for this s-step (8x 2 vector loads) ----
        v16h ka[2][2];
#pragma unroll
        for (int sub = 0; sub < 2; ++sub)
#pragma unroll
            for (int ke = 0; ke < 2; ++ke)
                ka[sub][ke] = load_a_f16(kb, ND, s0 + 16 * sub, ke * 32, lane);
        v16h vf[4];
#pragma unroll
        for (int et = 0; et < 4; ++et)
            vf[et] = load_a_f16(vtb, NS, et * 16, s0, lane);

        // ---- S^T = K @ Q^T for two 16-row sub-tiles ----
        float p0[8], p1[8];
#pragma unroll
        for (int sub = 0; sub < 2; ++sub) {
            v8f c = vzero8();
            c = wmma32(ka[sub][0], qf0, c);
            c = wmma32(ka[sub][1], qf1, c);
#pragma unroll
            for (int r = 0; r < 8; ++r) {
                int sg = s0 + 16 * sub + rsel + r;
                float val = c[r] * 0.125f;                 // 1/sqrt(64)
                if (sg > tcol) val = -1e30f;               // causal mask
                if (sub == 0) p0[r] = val; else p1[r] = val;
            }
        }
        // ---- online softmax (lane pair via xor-16 shuffle) ----
        float mx = -1e30f;
#pragma unroll
        for (int r = 0; r < 8; ++r) mx = fmaxf(mx, fmaxf(p0[r], p1[r]));
        mx = fmaxf(mx, __shfl_xor(mx, 16, 32));
        float mnew  = fmaxf(mrun, mx);
        float alpha = expf(mrun - mnew);
        float lsum  = 0.0f;
#pragma unroll
        for (int r = 0; r < 8; ++r) {
            p0[r] = expf(p0[r] - mnew);
            p1[r] = expf(p1[r] - mnew);
            lsum += p0[r] + p1[r];
        }
        lsum += __shfl_xor(lsum, 16, 32);
        lrun = lrun * alpha + lsum;
        mrun = mnew;
#pragma unroll
        for (int et = 0; et < 4; ++et)
#pragma unroll
            for (int r = 0; r < 8; ++r) oacc[et][r] *= alpha;

        // ---- repack P^T (two 16x16 C frags) into one B frag (K=32) ----
        v16h pf;
#pragma unroll
        for (int j = 0; j < 8; ++j) {
            float x0 = __shfl_xor(p0[j], 16, 32);
            float x1 = __shfl_xor(p1[j], 16, 32);
            if (lane < 16) {
                pf[j]     = (_Float16)p0[j];
                pf[8 + j] = (_Float16)x0;
            } else {
                pf[j]     = (_Float16)x1;
                pf[8 + j] = (_Float16)p1[j];
            }
        }
        // ---- O^T += V^T @ P^T ----
#pragma unroll
        for (int et = 0; et < 4; ++et)
            oacc[et] = wmma32(vf[et], pf, oacc[et]);
    }

    float inv = 1.0f / lrun;
#pragma unroll
    for (int et = 0; et < 4; ++et)
#pragma unroll
        for (int r = 0; r < 8; ++r) {
            int e = et * 16 + rsel + r;
            o[rowbase + (size_t)tcol * ND + hbase + e] = (_Float16)(oacc[et][r] * inv);
        }
}

// ---------------------------------------------------------------------------
// LayerNorm (fp32 in -> f16 out), one block per row of 768.
// ---------------------------------------------------------------------------
__global__ __launch_bounds__(256) void ln_kernel(
    const float* __restrict__ x, const float* __restrict__ w,
    const float* __restrict__ b, _Float16* __restrict__ out) {
    __shared__ float s1[256], s2[256];
    int row = blockIdx.x, tid = threadIdx.x;
    const float* xr = x + (size_t)row * ND;
    float a0 = xr[tid], a1 = xr[tid + 256], a2 = xr[tid + 512];
    s1[tid] = a0 + a1 + a2;
    s2[tid] = a0 * a0 + a1 * a1 + a2 * a2;
    __syncthreads();
    for (int st = 128; st > 0; st >>= 1) {
        if (tid < st) { s1[tid] += s1[tid + st]; s2[tid] += s2[tid + st]; }
        __syncthreads();
    }
    float mu   = s1[0] * (1.0f / ND);
    float var  = s2[0] * (1.0f / ND) - mu * mu;
    float rstd = rsqrtf(var + 1e-5f);
    _Float16* orow = out + (size_t)row * ND;
    orow[tid]       = (_Float16)((a0 - mu) * rstd * w[tid]       + b[tid]);
    orow[tid + 256] = (_Float16)((a1 - mu) * rstd * w[tid + 256] + b[tid + 256]);
    orow[tid + 512] = (_Float16)((a2 - mu) * rstd * w[tid + 512] + b[tid + 512]);
}

// ---------------------------------------------------------------------------
// Embedding: x = wte[idx] + wpe[t]
// ---------------------------------------------------------------------------
__global__ __launch_bounds__(256) void embed_kernel(
    const int* __restrict__ idx, const float* __restrict__ wte,
    const float* __restrict__ wpe, float* __restrict__ x) {
    int row = blockIdx.x;
    int t   = row & (NS - 1);
    int tok = idx[row];
    const float* wr = wte + (size_t)tok * ND;
    const float* pr = wpe + (size_t)t * ND;
    float* xr = x + (size_t)row * ND;
    for (int c = threadIdx.x; c < ND; c += 256) xr[c] = wr[c] + pr[c];
}

// ---------------------------------------------------------------------------
// Loss: cross-entropy with log-softmax, one block per row, atomicAdd of mean.
// ---------------------------------------------------------------------------
__global__ void init_loss_kernel(float* loss) { *loss = 0.0f; }

__global__ __launch_bounds__(256) void loss_kernel(
    const float* __restrict__ logits, const int* __restrict__ tgt,
    float* __restrict__ loss) {
    __shared__ float sh[256];
    int row = blockIdx.x, tid = threadIdx.x;
    const float* lr = logits + (size_t)row * NV;
    float mx = -3.0e38f;
    for (int c = tid; c < NV; c += 256) mx = fmaxf(mx, lr[c]);
    sh[tid] = mx; __syncthreads();
    for (int st = 128; st > 0; st >>= 1) {
        if (tid < st) sh[tid] = fmaxf(sh[tid], sh[tid + st]);
        __syncthreads();
    }
    mx = sh[0]; __syncthreads();
    float sum = 0.0f;
    for (int c = tid; c < NV; c += 256) sum += expf(lr[c] - mx);
    sh[tid] = sum; __syncthreads();
    for (int st = 128; st > 0; st >>= 1) {
        if (tid < st) sh[tid] += sh[tid + st];
        __syncthreads();
    }
    if (tid == 0) {
        float lp = lr[tgt[row]] - mx - logf(sh[0]);
        atomicAdd(loss, -lp * (1.0f / (float)NM));
    }
}

// ---------------------------------------------------------------------------
extern "C" void kernel_launch(void* const* d_in, const int* in_sizes, int n_in,
                              void* d_out, int out_size, void* d_ws, size_t ws_size,
                              hipStream_t stream) {
    (void)in_sizes; (void)n_in; (void)out_size; (void)ws_size;
    const int*   idx     = (const int*)d_in[0];
    const int*   targets = (const int*)d_in[1];
    const float* wte  = (const float*)d_in[2];
    const float* wpe  = (const float*)d_in[3];
    const float* Wq   = (const float*)d_in[4];
    const float* bq   = (const float*)d_in[5];
    const float* Wk   = (const float*)d_in[6];
    const float* bk   = (const float*)d_in[7];
    const float* Wv   = (const float*)d_in[8];
    const float* bv   = (const float*)d_in[9];
    const float* Wo   = (const float*)d_in[10];
    const float* bo   = (const float*)d_in[11];
    const float* ln_w = (const float*)d_in[12];
    const float* ln_b = (const float*)d_in[13];
    const float* W1   = (const float*)d_in[14];
    const float* b1   = (const float*)d_in[15];
    const float* W2   = (const float*)d_in[16];
    const float* b2   = (const float*)d_in[17];
    const float* lnf_w = (const float*)d_in[18];
    const float* lnf_b = (const float*)d_in[19];

    // Workspace carve-up (256B aligned)
    char* ws = (char*)d_ws;
    size_t off = 0;
    auto carve = [&](size_t bytes) {
        char* p = ws + off;
        off = (off + bytes + 255) & ~(size_t)255;
        return p;
    };
    float*    x    = (float*)   carve((size_t)NM * ND * sizeof(float));
    _Float16* h16  = (_Float16*)carve((size_t)NM * ND * sizeof(_Float16));
    _Float16* q16  = (_Float16*)carve((size_t)NM * ND * sizeof(_Float16));
    _Float16* k16  = (_Float16*)carve((size_t)NM * ND * sizeof(_Float16));
    _Float16* vt16 = (_Float16*)carve((size_t)NM * ND * sizeof(_Float16));
    _Float16* o16  = (_Float16*)carve((size_t)NM * ND * sizeof(_Float16));
    _Float16* g16  = (_Float16*)carve((size_t)NM * NFF * sizeof(_Float16));
    _Float16* wq_t = (_Float16*)carve((size_t)ND * ND * sizeof(_Float16));
    _Float16* wk_t = (_Float16*)carve((size_t)ND * ND * sizeof(_Float16));
    _Float16* wv_t = (_Float16*)carve((size_t)ND * ND * sizeof(_Float16));
    _Float16* wo_t = (_Float16*)carve((size_t)ND * ND * sizeof(_Float16));
    _Float16* w1_t = (_Float16*)carve((size_t)NFF * ND * sizeof(_Float16)); // [3072][768]
    _Float16* w2_t = (_Float16*)carve((size_t)ND * NFF * sizeof(_Float16)); // [768][3072]
    _Float16* wte16 = (_Float16*)carve((size_t)NVP * ND * sizeof(_Float16)); // padded

    float* logits = (float*)d_out;
    float* loss   = logits + (size_t)NM * NV;

    dim3 blk(256);
    dim3 gD(NM / 64, ND / 256);    // N = 768
    dim3 gF(NM / 64, NFF / 256);   // N = 3072
    dim3 gV(NM / 64, (NV + 255) / 256);
    dim3 gTd(ND / 32, ND / 32);    // 768x768 transpose
    dim3 gT1(ND / 32, NFF / 32);   // K=768,  N=3072
    dim3 gT2(NFF / 32, ND / 32);   // K=3072, N=768

    embed_kernel<<<NM, blk, 0, stream>>>(idx, wte, wpe, x);
    f32_to_f16_kernel<<<(int)(((long)NV * ND + 1023) / 1024), blk, 0, stream>>>(
        wte, wte16, (long)NV * ND);

    for (int l = 0; l < NL; ++l) {
        const float* lw = ln_w + (size_t)l * ND;
        const float* lb = ln_b + (size_t)l * ND;
        size_t wqkv = (size_t)l * NH * ND * NHS;     // = l*589824

        // Stage this layer's weights as f16 [N,K]
        transpose_w_kernel<1><<<gTd, blk, 0, stream>>>(Wq + wqkv, wq_t, ND, ND);
        transpose_w_kernel<1><<<gTd, blk, 0, stream>>>(Wk + wqkv, wk_t, ND, ND);
        transpose_w_kernel<1><<<gTd, blk, 0, stream>>>(Wv + wqkv, wv_t, ND, ND);
        transpose_w_kernel<0><<<gTd, blk, 0, stream>>>(Wo + (size_t)l * ND * ND, wo_t, ND, ND);
        transpose_w_kernel<0><<<gT1, blk, 0, stream>>>(W1 + (size_t)l * ND * NFF, w1_t, ND, NFF);
        transpose_w_kernel<0><<<gT2, blk, 0, stream>>>(W2 + (size_t)l * NFF * ND, w2_t, NFF, ND);

        ln_kernel<<<NM, blk, 0, stream>>>(x, lw, lb, h16);

        gemm_kernel<EPI_F16><<<gD, blk, 0, stream>>>(
            h16, ND, wq_t, bq + (size_t)l * ND, nullptr, nullptr, q16, ND, NM, ND, ND);
        gemm_kernel<EPI_F16><<<gD, blk, 0, stream>>>(
            h16, ND, wk_t, bk + (size_t)l * ND, nullptr, nullptr, k16, ND, NM, ND, ND);
        gemm_kernel<EPI_F16T><<<gD, blk, 0, stream>>>(
            h16, ND, wv_t, bv + (size_t)l * ND, nullptr, nullptr, vt16, ND, NM, ND, ND);

        attn_kernel<<<NB * NH * (NS / 128), blk, 0, stream>>>(q16, k16, vt16, o16);

        gemm_kernel<EPI_RES_F32><<<gD, blk, 0, stream>>>(
            o16, ND, wo_t, bo + (size_t)l * ND, x, x, nullptr, ND, NM, ND, ND);

        ln_kernel<<<NM, blk, 0, stream>>>(x, lw, lb, h16);

        gemm_kernel<EPI_GELU_F16><<<gF, blk, 0, stream>>>(
            h16, ND, w1_t, b1 + (size_t)l * NFF, nullptr, nullptr, g16, NFF, NM, NFF, ND);

        gemm_kernel<EPI_RES_F32><<<gD, blk, 0, stream>>>(
            g16, NFF, w2_t, b2 + (size_t)l * ND, x, x, nullptr, ND, NM, ND, NFF);
    }

    ln_kernel<<<NM, blk, 0, stream>>>(x, lnf_w, lnf_b, h16);

    // logits = h @ wte^T  (tied head; wte16 is [V,D] = [N,K], padded rows
    // beyond V are never stored thanks to the epilogue col<N mask)
    gemm_kernel<EPI_F32><<<gV, blk, 0, stream>>>(
        h16, ND, wte16, nullptr, nullptr, logits, nullptr, NV, NM, NV, ND);

    init_loss_kernel<<<1, 1, 0, stream>>>(loss);
    loss_kernel<<<NM, blk, 0, stream>>>(logits, targets, loss);
}
